// AttentionHeader_65472481460778
// MI455X (gfx1250) — matrix operand order repaired and checked by reading, not compiled, not run on hardware
//
#include <hip/hip_runtime.h>
#include <hip/hip_bf16.h>

// ---------------------------------------------------------------------------
// Types for CDNA5 WMMA bf16 path
// ---------------------------------------------------------------------------
typedef __bf16 bf16_t;
typedef __attribute__((ext_vector_type(16))) __bf16 bf16x16;
typedef __attribute__((ext_vector_type(8)))  __bf16 bf16x8;
typedef __attribute__((ext_vector_type(8)))  float  f32x8;
typedef int v4i_ __attribute__((vector_size(16)));   // matches builtin param type

#define AS1 __attribute__((address_space(1)))
#define AS3 __attribute__((address_space(3)))

#define B_   2
#define S_   1024
#define H_   1024
#define NH_  16
#define HD_  64
#define MP_  128
#define NROWS_ (B_ * S_)          // 2048 rows of X
#define NQROWS_ (B_ * NH_ * S_)   // 32768 flattened (b,h,s) rows
#define LDB_ 40                   // LDS leading dim: 80B rows, 16B aligned, bank-conflict free

#if __has_builtin(__builtin_amdgcn_global_load_async_to_lds_b128)
#define HAVE_ASYNC_LDS 1
#endif

__device__ __forceinline__ void wait_async_zero() {
#if defined(HAVE_ASYNC_LDS)
#if __has_builtin(__builtin_amdgcn_s_wait_asynccnt)
    __builtin_amdgcn_s_wait_asynccnt(0);
#else
    asm volatile("s_wait_asynccnt 0" ::: "memory");
#endif
#endif
}

__device__ __forceinline__ f32x8 wmma_bf16(bf16x16 a, bf16x16 b, f32x8 c) {
    // D = A(16x32 bf16) * B(32x16 bf16) + C(16x16 f32)
    return __builtin_amdgcn_wmma_f32_16x16x32_bf16(
        false, a, false, b, (short)0, c, false, false);
}

union FragU { bf16x16 v; bf16x8 h[2]; };

// A fragment: 16x32 bf16, source row-major, ld multiple of 8 (16B-aligned rows).
// Lane L: row m = L&15, half h = L>>4; elems [0..7] = r[h*8 ..], [8..15] = r[16+h*8 ..]
__device__ __forceinline__ bf16x16 load_a(const bf16_t* base, int ld, int lane) {
    const int m = lane & 15, h = lane >> 4;
    const bf16_t* r = base + m * ld + h * 8;
    FragU u;
    u.h[0] = *(const bf16x8*)(r);
    u.h[1] = *(const bf16x8*)(r + 16);
    return u.v;
}

// B fragment: 32x16 (KxN) bf16, source "N-major, K-contiguous": (k,n) at base[n*ld+k].
// Lane L: col n = L&15, half h = L>>4; elems [0..7] = K h*8.., [8..15] = K 16+h*8..
__device__ __forceinline__ bf16x16 load_b(const bf16_t* base, int ld, int lane) {
    const int n = lane & 15, h = lane >> 4;
    const bf16_t* c = base + n * ld + h * 8;
    FragU u;
    u.h[0] = *(const bf16x8*)(c);
    u.h[1] = *(const bf16x8*)(c + 16);
    return u.v;
}

// ---------------------------------------------------------------------------
// Kernel 1: f32 -> bf16 cast (8 elems/thread), zero-padded to n_dst
// ---------------------------------------------------------------------------
__global__ void cast_f32_bf16(const float* __restrict__ src,
                              bf16_t* __restrict__ dst, int n_src, int n_dst) {
    const int i0 = (blockIdx.x * 256 + threadIdx.x) * 8;
    if (i0 >= n_dst) return;
#pragma unroll
    for (int j = 0; j < 8; ++j) {
        const int i = i0 + j;
        dst[i] = (i < n_src) ? (bf16_t)src[i] : (bf16_t)0.0f;
    }
}

// ---------------------------------------------------------------------------
// Kernel 2: projection GEMM  out = split_heads(X @ W + bias)   (bf16 out)
//   Workgroup: 128 thr (4 waves); C-tile 32(M) x 64(N); K chunk 32.
// ---------------------------------------------------------------------------
__global__ void qkv_gemm(const bf16_t* __restrict__ X,
                         const bf16_t* __restrict__ W,
                         const float*  __restrict__ bias,
                         bf16_t* __restrict__ out) {
    __shared__ bf16_t As[32][32];
    __shared__ bf16_t BT[64][LDB_];   // transposed: BT[n][k]

    const int tid  = threadIdx.x;
    const int lane = tid & 31;
    const int wave = tid >> 5;
    const int mbase = blockIdx.x * 32;
    const int nbase = blockIdx.y * 64;

    f32x8 c0 = {}, c1 = {};

    for (int kc = 0; kc < H_ / 32; ++kc) {
        const int k0 = kc * 32;
        // stage A tile 32x32: one b128 per thread (async to LDS when available)
        {
            const int row = tid >> 2, col = (tid & 3) * 8;
            const bf16_t* gsrc = X + (size_t)(mbase + row) * H_ + k0 + col;
#if defined(HAVE_ASYNC_LDS)
            __builtin_amdgcn_global_load_async_to_lds_b128(
                (AS1 v4i_*)gsrc, (AS3 v4i_*)&As[row][col], 0, 0);
#else
            *(bf16x8*)&As[row][col] = *(const bf16x8*)gsrc;
#endif
        }
        // stage W tile 32(k)x64(n): 2 b128 reads + transposed scatter BT[n][k]
        {
            const int k = tid >> 2, n16 = (tid & 3) * 16;
            const bf16_t* wsrc = W + (size_t)(k0 + k) * H_ + nbase + n16;
            bf16x8 w0 = *(const bf16x8*)(wsrc);
            bf16x8 w1 = *(const bf16x8*)(wsrc + 8);
#pragma unroll
            for (int j = 0; j < 8; ++j) {
                BT[n16 + j][k]     = w0[j];
                BT[n16 + 8 + j][k] = w1[j];
            }
        }
        wait_async_zero();
        __syncthreads();

        bf16x16 bf = load_b(&BT[wave * 16][0], LDB_, lane);
        bf16x16 a0 = load_a(&As[0][0], 32, lane);
        bf16x16 a1 = load_a(&As[16][0], 32, lane);
        c0 = wmma_bf16(a0, bf, c0);
        c1 = wmma_bf16(a1, bf, c1);
        __syncthreads();
    }

    // store with head split: row m -> (b,s); col n -> (h,d)
    const int nl = lane & 15, half = lane >> 4;
    const int n = nbase + wave * 16 + nl;
    const float bv = bias[n];
    const int h = n >> 6, d = n & 63;
#pragma unroll
    for (int g = 0; g < 8; ++g) {
        {
            const int m = mbase + g + 8 * half;
            const int b = m >> 10, s = m & 1023;
            out[(((b * NH_) + h) * S_ + s) * HD_ + d] = (bf16_t)(c0[g] + bv);
        }
        {
            const int m = mbase + 16 + g + 8 * half;
            const int b = m >> 10, s = m & 1023;
            out[(((b * NH_) + h) * S_ + s) * HD_ + d] = (bf16_t)(c1[g] + bv);
        }
    }
}

// ---------------------------------------------------------------------------
// Kernel 3: RQ[row, j] = sum_d Q[row,d] * dist_emb[j,d]   (f32 out, j<256)
// ---------------------------------------------------------------------------
__global__ void rq_gemm(const bf16_t* __restrict__ Qb,
                        const bf16_t* __restrict__ de,   // [256][64] bf16
                        float* __restrict__ RQ) {
    const int tid  = threadIdx.x;
    const int lane = tid & 31;
    const int wave = tid >> 5;
    const int rowbase = blockIdx.x * 16;

    const bf16_t* qp = Qb + (size_t)rowbase * HD_;
    bf16x16 a0 = load_a(qp, HD_, lane);        // d = 0..31
    bf16x16 a1 = load_a(qp + 32, HD_, lane);   // d = 32..63

    const int nl = lane & 15, half = lane >> 4;
#pragma unroll
    for (int t = 0; t < 4; ++t) {
        const int j0 = (wave * 4 + t) * 16;
        bf16x16 b0 = load_b(de + j0 * HD_, HD_, lane);
        bf16x16 b1 = load_b(de + j0 * HD_ + 32, HD_, lane);
        f32x8 c = {};
        c = wmma_bf16(a0, b0, c);
        c = wmma_bf16(a1, b1, c);
#pragma unroll
        for (int g = 0; g < 8; ++g) {
            const int m = rowbase + g + 8 * half;
            RQ[(size_t)m * 256 + j0 + nl] = c[g];
        }
    }
}

// ---------------------------------------------------------------------------
// Kernel 4: fused flash attention with relative_key bias gather.
//   Grid: (S/64, B*NH). Block: 128 thr (4 waves, 16 query rows each).
// ---------------------------------------------------------------------------
__global__ void attn_kernel(const bf16_t* __restrict__ Qb,
                            const bf16_t* __restrict__ Kb,
                            const bf16_t* __restrict__ Vb,
                            const float*  __restrict__ RQ,
                            float* __restrict__ out) {
    __shared__ bf16_t VT[64][LDB_];    // VT[d][key]  (key tile of 32)
    __shared__ bf16_t Pb[4][16][32];   // per-wave probs, row-major

    const int tid  = threadIdx.x;
    const int lane = tid & 31;
    const int wave = tid >> 5;
    const int bh   = blockIdx.y;                 // b*NH + h
    const int qb   = blockIdx.x * 64;
    const int qrow0 = qb + wave * 16;

    const bf16_t* Qp = Qb + ((size_t)bh * S_ + qrow0) * HD_;
    const bf16_t* Kp = Kb + (size_t)bh * S_ * HD_;
    const bf16_t* Vp = Vb + (size_t)bh * S_ * HD_;
    const float*  rqp = RQ + ((size_t)bh * S_ + qrow0) * 256;

    bf16x16 aq0 = load_a(Qp, HD_, lane);
    bf16x16 aq1 = load_a(Qp + 32, HD_, lane);

    f32x8 o0 = {}, o1 = {}, o2 = {}, o3 = {};
    float mrow[8], lrow[8];
#pragma unroll
    for (int g = 0; g < 8; ++g) { mrow[g] = -3.0e38f; lrow[g] = 0.0f; }

    const int nl = lane & 15, half = lane >> 4;

    for (int kt = 0; kt < S_ / 32; ++kt) {
        const int kb = kt * 32;

        // prefetch next K/V tiles (lowers to global_prefetch_b8)
        if (kb + 32 < S_) {
            __builtin_prefetch(Kp + (size_t)(kb + 32 + (tid >> 2)) * HD_ + (tid & 3) * 16, 0, 1);
            __builtin_prefetch(Vp + (size_t)(kb + 32 + (tid >> 2)) * HD_ + (tid & 3) * 16, 0, 1);
        }

        // stage V tile transposed: b128 reads, transposed scatter into VT[d][key]
#pragma unroll
        for (int e = 0; e < 2; ++e) {
            const int idx = e * 128 + tid;     // 256 chunks of 8
            const int key = idx >> 3;          // 0..31
            const int d8  = (idx & 7) * 8;
            bf16x8 v = *(const bf16x8*)(Vp + (size_t)(kb + key) * HD_ + d8);
#pragma unroll
            for (int j = 0; j < 8; ++j) VT[d8 + j][key] = v[j];
        }
        __syncthreads();

        // ---- scores: two 16-col tiles, QK^T over HD=64 (2 WMMAs each) ----
        f32x8 sc[2];
#pragma unroll
        for (int t = 0; t < 2; ++t) {
            const bf16_t* kbase = Kp + (size_t)(kb + t * 16) * HD_;
            bf16x16 b0 = load_b(kbase, HD_, lane);
            bf16x16 b1 = load_b(kbase + 32, HD_, lane);
            f32x8 c = {};
            c = wmma_bf16(aq0, b0, c);
            c = wmma_bf16(aq1, b1, c);
            // relative-position bias gather + scale
            const int r = kb + t * 16 + nl;
#pragma unroll
            for (int g = 0; g < 8; ++g) {
                const int mloc = g + 8 * half;
                const int l = qrow0 + mloc;
                int dd = l - r;
                dd = (dd > MP_ - 1) ? (MP_ - 1) : dd;
                int idx = dd + MP_ - 1;
                idx = (idx < 0) ? 0 : idx;
                if (l == 0 || r == 0) idx = MP_ - 1;
                const float bias = rqp[(size_t)mloc * 256 + idx];
                c[g] = (c[g] + bias) * 0.125f;   // 1/sqrt(64)
            }
            sc[t] = c;
        }

        // ---- online softmax (reduce across the 16 lanes of each half) ----
#pragma unroll
        for (int g = 0; g < 8; ++g) {
            float v = fmaxf(sc[0][g], sc[1][g]);
            v = fmaxf(v, __shfl_xor(v, 1, 32));
            v = fmaxf(v, __shfl_xor(v, 2, 32));
            v = fmaxf(v, __shfl_xor(v, 4, 32));
            v = fmaxf(v, __shfl_xor(v, 8, 32));
            const float mnew = fmaxf(mrow[g], v);
            const float alpha = __expf(mrow[g] - mnew);
            mrow[g] = mnew;
            const float p0 = __expf(sc[0][g] - mnew);
            const float p1 = __expf(sc[1][g] - mnew);
            sc[0][g] = p0; sc[1][g] = p1;
            float s = p0 + p1;
            s += __shfl_xor(s, 1, 32);
            s += __shfl_xor(s, 2, 32);
            s += __shfl_xor(s, 4, 32);
            s += __shfl_xor(s, 8, 32);
            lrow[g] = lrow[g] * alpha + s;
            o0[g] *= alpha; o1[g] *= alpha; o2[g] *= alpha; o3[g] *= alpha;
        }

        // ---- P (C layout) -> per-wave LDS row-major bf16 ----
#pragma unroll
        for (int g = 0; g < 8; ++g) {
            const int m = g + 8 * half;
            Pb[wave][m][nl]      = (bf16_t)sc[0][g];
            Pb[wave][m][16 + nl] = (bf16_t)sc[1][g];
        }
        asm volatile("s_wait_dscnt 0" ::: "memory");   // CDNA5 split DS counter
        __builtin_amdgcn_wave_barrier();

        bf16x16 ap = load_a(&Pb[wave][0][0], 32, lane);

        // ---- ctx += P(16x32) * V(32x64): 4 B-tiles of V from VT ----
        {
            bf16x16 bv0 = load_b(&VT[0][0],  LDB_, lane);
            bf16x16 bv1 = load_b(&VT[16][0], LDB_, lane);
            bf16x16 bv2 = load_b(&VT[32][0], LDB_, lane);
            bf16x16 bv3 = load_b(&VT[48][0], LDB_, lane);
            o0 = wmma_bf16(ap, bv0, o0);
            o1 = wmma_bf16(ap, bv1, o1);
            o2 = wmma_bf16(ap, bv2, o2);
            o3 = wmma_bf16(ap, bv3, o3);
        }
        __syncthreads();   // protect VT before next tile overwrites it
    }

    // ---- finalize: divide by l, store [B,S,H] f32 ----
    const int b = bh >> 4, h = bh & 15;
#pragma unroll
    for (int g = 0; g < 8; ++g) {
        const float inv = 1.0f / lrow[g];
        const int s = qrow0 + g + 8 * half;
        float* op = out + ((size_t)(b * S_ + s) * H_) + h * HD_;
        op[0 * 16 + nl] = o0[g] * inv;
        op[1 * 16 + nl] = o1[g] * inv;
        op[2 * 16 + nl] = o2[g] * inv;
        op[3 * 16 + nl] = o3[g] * inv;
    }
}

// ---------------------------------------------------------------------------
// Host-side launch
// ---------------------------------------------------------------------------
extern "C" void kernel_launch(void* const* d_in, const int* in_sizes, int n_in,
                              void* d_out, int out_size, void* d_ws, size_t ws_size,
                              hipStream_t stream) {
    (void)in_sizes; (void)n_in; (void)out_size; (void)ws_size;

    const float* hidden = (const float*)d_in[0];
    const float* Wq = (const float*)d_in[1];
    const float* bq = (const float*)d_in[2];
    const float* Wk = (const float*)d_in[3];
    const float* bk = (const float*)d_in[4];
    const float* Wv = (const float*)d_in[5];
    const float* bv = (const float*)d_in[6];
    const float* de = (const float*)d_in[7];
    float* out = (float*)d_out;

    // workspace carve-up (bytes)
    char* ws = (char*)d_ws;
    const size_t szX  = (size_t)NROWS_ * H_ * 2;        // 4 MiB hidden bf16
    const size_t szW  = (size_t)H_ * H_ * 2;            // 2 MiB each
    const size_t szDE = (size_t)256 * HD_ * 2;          // 32 KiB padded
    const size_t szQ  = (size_t)NQROWS_ * HD_ * 2;      // 4 MiB each

    bf16_t* hb  = (bf16_t*)(ws);
    bf16_t* wqb = (bf16_t*)(ws + szX);
    bf16_t* wkb = (bf16_t*)(ws + szX + szW);
    bf16_t* wvb = (bf16_t*)(ws + szX + 2 * szW);
    bf16_t* deb = (bf16_t*)(ws + szX + 3 * szW);
    bf16_t* Qb  = (bf16_t*)(ws + szX + 3 * szW + szDE);
    bf16_t* Kbp = (bf16_t*)(ws + szX + 3 * szW + szDE + szQ);
    bf16_t* Vbp = (bf16_t*)(ws + szX + 3 * szW + szDE + 2 * szQ);
    float*  RQ  = (float*) (ws + szX + 3 * szW + szDE + 3 * szQ);  // 32 MiB

    // 1) casts (8 elems/thread)
    {
        int n = NROWS_ * H_;
        cast_f32_bf16<<<(n / 8 + 255) / 256, 256, 0, stream>>>(hidden, hb, n, n);
        n = H_ * H_;
        cast_f32_bf16<<<(n / 8 + 255) / 256, 256, 0, stream>>>(Wq, wqb, n, n);
        cast_f32_bf16<<<(n / 8 + 255) / 256, 256, 0, stream>>>(Wk, wkb, n, n);
        cast_f32_bf16<<<(n / 8 + 255) / 256, 256, 0, stream>>>(Wv, wvb, n, n);
        cast_f32_bf16<<<(256 * HD_ / 8 + 255) / 256, 256, 0, stream>>>(
            de, deb, (2 * MP_ - 1) * HD_, 256 * HD_);
    }

    // 2) Q/K/V projections (bf16 WMMA, head-split outputs)
    {
        dim3 grid(NROWS_ / 32, H_ / 64);
        qkv_gemm<<<grid, 128, 0, stream>>>(hb, wqb, bq, Qb);
        qkv_gemm<<<grid, 128, 0, stream>>>(hb, wkb, bk, Kbp);
        qkv_gemm<<<grid, 128, 0, stream>>>(hb, wvb, bv, Vbp);
    }

    // 3) RQ = Q . dist_emb^T  (per-row dot with all 255 distance embeddings)
    rq_gemm<<<NQROWS_ / 16, 128, 0, stream>>>(Qb, deb, RQ);

    // 4) fused flash attention with bias gather
    {
        dim3 grid(S_ / 64, B_ * NH_);
        attn_kernel<<<grid, 128, 0, stream>>>(Qb, Kbp, Vbp, RQ, out);
    }
}
